// OutputLayer_33509334843944
// MI455X (gfx1250) — matrix operand order repaired and checked by reading, not compiled, not run on hardware
//
#include <hip/hip_runtime.h>
#include <hip/hip_bf16.h>
#include <math.h>

// ---------------------------------------------------------------------------
// MatchingLayer pipeline for MI455X (gfx1250, wave32, WMMA)
//   scores[r,n] = (hs@W)[r,:] . es[n,:]   via v_wmma_f32_16x16x32_bf16
//   then normalize / dual log-softmax / masked argmax / top-k / gather.
// ---------------------------------------------------------------------------

typedef __attribute__((ext_vector_type(16))) __bf16 v16bf;
typedef __attribute__((ext_vector_type(8)))  float  v8f;

#define RROWS 64
#define DDIM  256
#define NEGV  (-1.0e9f)
#define APITCH 264   // ushorts per LDS A row: stride = 132 dwords == 4 (mod 64 banks)

static __device__ __forceinline__ unsigned short f2bf_rne(float f) {
  unsigned u = __float_as_uint(f);
  unsigned r = u + 0x7fffu + ((u >> 16) & 1u);   // round-to-nearest-even
  return (unsigned short)(r >> 16);
}

// Fast pack for the GEMM hot loop: round-half-up to bf16, all 32-bit ALU.
// ~3 VALU ops/dword (add, add, shift+mask-or -> v_perm-able), no b16 moves.
static __device__ __forceinline__ unsigned pack2bf_fast(float a, float b) {
  const unsigned ua = __float_as_uint(a) + 0x8000u;
  const unsigned ub = __float_as_uint(b) + 0x8000u;
  return (ua >> 16) | (ub & 0xffff0000u);
}

// --- init: zero the f64 accumulators (ws is poisoned, not zeroed) ----------
__global__ void k_init(double* __restrict__ sums) {
  sums[0] = 0.0; sums[1] = 0.0;
}

// --- K0: hsW = hs @ W (f32 accurate), stored as bf16 -----------------------
__global__ void k0_hsw(const float* __restrict__ hs, const float* __restrict__ W,
                       unsigned short* __restrict__ hswbf) {
  __shared__ float srow[DDIM];
  const int m = blockIdx.x;       // 64 blocks
  const int j = threadIdx.x;      // 256 threads = one output column each
  srow[j] = hs[m * DDIM + j];
  __syncthreads();
  float acc = 0.f;
  #pragma unroll 8
  for (int k = 0; k < DDIM; ++k) acc = fmaf(srow[k], W[k * DDIM + j], acc);
  hswbf[m * DDIM + j] = f2bf_rne(acc);
}

// --- K1: scores = hsW @ es^T  (bf16 WMMA, f32 accum) + sum/sumsq -----------
__global__ void __launch_bounds__(256)
k1_gemm(const float* __restrict__ es, const unsigned short* __restrict__ hswbf,
        float* __restrict__ scores, double* __restrict__ sums, int N) {
  __shared__ unsigned short ldsA[RROWS * APITCH];   // padded, bank-conflict-free
  __shared__ double redS[256], redQ[256];
  const int tid = threadIdx.x;

  // cooperative fill of A into padded LDS (2048 uint4 total)
  {
    const uint4* src = (const uint4*)hswbf;
    for (int i = tid; i < RROWS * 32; i += 256) {
      const int m = i >> 5, c = i & 31;
      *(uint4*)&ldsA[m * APITCH + c * 8] = src[i];
    }
  }
  __syncthreads();

  const int wave = tid >> 5;
  const int lane = tid & 31;
  const int lm   = lane & 15;      // matrix col (B) / row (A) within tile
  const int h    = lane >> 4;      // half-wave selector

  int n = blockIdx.x * 128 + wave * 16 + lm;     // this lane's es row / C col
  const bool nok = (n < N);
  if (!nok) n = N - 1;                            // clamp loads, predicate stores
  const float* __restrict__ esrow = es + (size_t)n * DDIM;

  const v8f zero8 = {0.f,0.f,0.f,0.f,0.f,0.f,0.f,0.f};
  v8f acc[4] = {zero8, zero8, zero8, zero8};

  #pragma unroll
  for (int kc = 0; kc < DDIM / 32; ++kc) {
    if (kc + 1 < DDIM / 32)
      __builtin_prefetch((const void*)(esrow + (kc + 1) * 32 + h * 16), 0, 3);

    // B fragment (32x16 K-major): lane holds K = kc*32 + h*16 .. +15 of col n.
    // Load all 64B first so the four b128 loads clause together.
    const float4* __restrict__ bp = (const float4*)(esrow + kc * 32 + h * 16);
    const float4 f0 = bp[0];
    const float4 f1 = bp[1];
    const float4 f2 = bp[2];
    const float4 f3 = bp[3];
    union { unsigned u[8]; v16bf v; } bf;
    bf.u[0] = pack2bf_fast(f0.x, f0.y);
    bf.u[1] = pack2bf_fast(f0.z, f0.w);
    bf.u[2] = pack2bf_fast(f1.x, f1.y);
    bf.u[3] = pack2bf_fast(f1.z, f1.w);
    bf.u[4] = pack2bf_fast(f2.x, f2.y);
    bf.u[5] = pack2bf_fast(f2.z, f2.w);
    bf.u[6] = pack2bf_fast(f3.x, f3.y);
    bf.u[7] = pack2bf_fast(f3.z, f3.w);

    // 4 A tiles (M = 0..63) from LDS, then 4 WMMAs
    #pragma unroll
    for (int mt = 0; mt < 4; ++mt) {
      union { unsigned u[8]; v16bf v; } af;
      const int rowOff = (mt * 16 + lm) * APITCH + kc * 32 + h * 16;
      *(uint4*)&af.u[0] = *(const uint4*)&ldsA[rowOff];        // K group 0
      *(uint4*)&af.u[4] = *(const uint4*)&ldsA[rowOff + 8];    // K group 1
      acc[mt] = __builtin_amdgcn_wmma_f32_16x16x32_bf16(
          false, af.v, false, bf.v, (short)0, acc[mt], false, false);
    }
  }

  // store C (lanes 0..15 / 16..31 give coalesced 64B segments per VGPR)
  // and accumulate mean/var statistics
  double s1 = 0.0, s2 = 0.0;
  #pragma unroll
  for (int mt = 0; mt < 4; ++mt) {
    #pragma unroll
    for (int i = 0; i < 8; ++i) {
      const float c = acc[mt][i];
      if (nok) {
        const int row = mt * 16 + h * 8 + i;
        scores[(size_t)row * N + n] = c;
        s1 += (double)c;
        s2 += (double)c * (double)c;
      }
    }
  }
  redS[tid] = s1; redQ[tid] = s2;
  __syncthreads();
  for (int off = 128; off > 0; off >>= 1) {
    if (tid < off) { redS[tid] += redS[tid + off]; redQ[tid] += redQ[tid + off]; }
    __syncthreads();
  }
  if (tid == 0) { atomicAdd(&sums[0], redS[0]); atomicAdd(&sums[1], redQ[0]); }
}

// --- K2: mu / 1/std (ddof=1) ----------------------------------------------
__global__ void k2_stats(const double* __restrict__ sums, float* __restrict__ stats,
                         int N) {
  const double M = (double)RROWS * (double)N;
  const double mu = sums[0] / M;
  double var = (sums[1] - sums[0] * sums[0] / M) / (M - 1.0);
  if (var < 1e-30) var = 1e-30;
  stats[0] = (float)mu;
  stats[1] = (float)(1.0 / sqrt(var));
}

// --- K3: per-column (over R=64): unmasked LSE + masked argmax --------------
__global__ void k3_col(const float* __restrict__ scores, const int* __restrict__ mask,
                       const float* __restrict__ stats, int* __restrict__ colIdx,
                       float* __restrict__ lseCol, int N) {
  const int n = blockIdx.x * blockDim.x + threadIdx.x;
  if (n >= N) return;
  const float mu = stats[0], istd = stats[1];
  float m = -INFINITY, s = 0.f;
  float bv = -INFINITY; int br = 0;
  #pragma unroll 4
  for (int r = 0; r < RROWS; ++r) {
    const float x = (scores[(size_t)r * N + n] - mu) * istd;
    if (x > m) { s = s * __expf(m - x) + 1.f; m = x; } else s += __expf(x - m);
    const float mv = mask[(size_t)r * N + n] ? x : NEGV;
    if (mv > bv) { bv = mv; br = r; }        // first-occurrence tie-break
  }
  colIdx[n] = br;
  lseCol[n] = m + __logf(s);
}

// --- K4: per-row (over N): unmasked LSE + masked (max,sumexp) --------------
__global__ void k4_row(const float* __restrict__ scores, const int* __restrict__ mask,
                       const float* __restrict__ stats, float* __restrict__ lseRow,
                       float* __restrict__ mMax, float* __restrict__ mSum, int N) {
  __shared__ float sm[256], ss[256], tm[256], ts[256];
  const int r = blockIdx.x, tid = threadIdx.x;
  const float mu = stats[0], istd = stats[1];
  float m = -INFINITY, s = 0.f, mm = -INFINITY, ms = 0.f;
  for (int n = tid; n < N; n += 256) {
    const float x = (scores[(size_t)r * N + n] - mu) * istd;
    if (x > m) { s = s * __expf(m - x) + 1.f; m = x; } else s += __expf(x - m);
    const float mv = mask[(size_t)r * N + n] ? x : NEGV;
    if (mv > mm) { ms = ms * __expf(mm - mv) + 1.f; mm = mv; } else ms += __expf(mv - mm);
  }
  sm[tid] = m; ss[tid] = s; tm[tid] = mm; ts[tid] = ms;
  __syncthreads();
  for (int off = 128; off > 0; off >>= 1) {
    if (tid < off) {
      float m2 = sm[tid + off], s2 = ss[tid + off];
      float mx = fmaxf(sm[tid], m2);
      ss[tid] = ss[tid] * __expf(sm[tid] - mx) + s2 * __expf(m2 - mx);
      sm[tid] = mx;
      m2 = tm[tid + off]; s2 = ts[tid + off];
      mx = fmaxf(tm[tid], m2);
      ts[tid] = ts[tid] * __expf(tm[tid] - mx) + s2 * __expf(m2 - mx);
      tm[tid] = mx;
    }
    __syncthreads();
  }
  if (tid == 0) { lseRow[r] = sm[0] + __logf(ss[0]); mMax[r] = tm[0]; mSum[r] = ts[0]; }
}

// --- K5: top-K per row by iterative strict-descending extraction -----------
// Ordering done in score domain with clamp mapped to x-space:
//   max(softmax,1e-6) ordering == ordering of max(x, mmax+log(1e-6*msum)).
__global__ void k5_topk(const float* __restrict__ scores, const int* __restrict__ mask,
                        const int* __restrict__ colIdx, const float* __restrict__ stats,
                        const float* __restrict__ mMax, const float* __restrict__ mSum,
                        int* __restrict__ topIdx, float* __restrict__ outExp,
                        int N, int K) {
  __shared__ float bv[256]; __shared__ int bn[256];
  __shared__ float prevV;   __shared__ int prevN;
  const int r = blockIdx.x, tid = threadIdx.x;
  const float mu = stats[0], istd = stats[1];
  const float mm = mMax[r];
  const float xc = mm + __logf(fmaxf(mSum[r], 1e-30f) * 1e-6f);  // clamp level
  if (tid == 0) { prevV = INFINITY; prevN = -1; }
  __syncthreads();
  for (int j = 0; j < K; ++j) {
    const float pv = prevV; const int pn = prevN;
    float lv = -INFINITY; int ln = 0x7fffffff;
    for (int n = tid; n < N; n += 256) {
      if (!mask[(size_t)r * N + n]) continue;
      if (colIdx[n] != r) continue;
      const float x = (scores[(size_t)r * N + n] - mu) * istd;
      const float v = fmaxf(x, xc);
      const bool below = (v < pv) || (v == pv && n > pn);   // strictly after prev
      if (below && (v > lv || (v == lv && n < ln))) { lv = v; ln = n; }
    }
    bv[tid] = lv; bn[tid] = ln;
    __syncthreads();
    for (int off = 128; off > 0; off >>= 1) {
      if (tid < off) {
        const float v2 = bv[tid + off]; const int n2 = bn[tid + off];
        if (v2 > bv[tid] || (v2 == bv[tid] && n2 < bn[tid])) { bv[tid] = v2; bn[tid] = n2; }
      }
      __syncthreads();
    }
    if (tid == 0) {
      const bool found = (bv[0] > -INFINITY);
      topIdx[r * K + j] = found ? bn[0] : -1;
      outExp[r * K + j] = found ? (float)bn[0] : -1.0f;     // expansions (as f32)
      prevV = found ? bv[0] : -INFINITY;
      prevN = found ? bn[0] : 0x7fffffff;
    }
    __syncthreads();
  }
}

// --- K6: exp_scores[r,j,i] = ok ? 2*x[i,nj] - lseRow[i] - lseCol[nj] : 0 ---
__global__ void k6_expsc(const float* __restrict__ scores, const float* __restrict__ stats,
                         const float* __restrict__ lseRow, const float* __restrict__ lseCol,
                         const int* __restrict__ topIdx, float* __restrict__ outE,
                         int N, int K) {
  const int idx = blockIdx.x * blockDim.x + threadIdx.x;
  const int total = RROWS * K * RROWS;
  if (idx >= total) return;
  const int i = idx & 63;
  const int j = (idx >> 6) % K;
  const int r = idx / (64 * K);
  const int nj = topIdx[r * K + j];
  float val = 0.f;
  if (nj >= 0) {
    const float x = (scores[(size_t)i * N + nj] - stats[0]) * stats[1];
    val = 2.f * x - lseRow[i] - lseCol[nj];
  }
  outE[idx] = val;
}

// ---------------------------------------------------------------------------
extern "C" void kernel_launch(void* const* d_in, const int* in_sizes, int n_in,
                              void* d_out, int out_size, void* d_ws, size_t ws_size,
                              hipStream_t stream) {
  const float* hs   = (const float*)d_in[0];
  const float* es   = (const float*)d_in[1];
  const float* W    = (const float*)d_in[2];
  const int*   mask = (const int*)d_in[3];
  // d_in[4] = n_output (device scalar); derive K deterministically from out_size:
  // out_size = R*K + R*K*R = R*K*(R+1)
  const int R = RROWS;
  const int N = in_sizes[1] / DDIM;
  int K = out_size / (R * (R + 1));
  if (K < 1) K = 1;

  char* ws = (char*)d_ws;
  double*         sums   = (double*)(ws + 0);              // 16 B
  float*          stats  = (float*)(ws + 16);              // mu, 1/std
  float*          lseRow = (float*)(ws + 32);              // 64 f32
  float*          mMaxA  = (float*)(ws + 32 + 256);
  float*          mSumA  = (float*)(ws + 32 + 512);
  int*            topIdx = (int*)(ws + 1024);              // up to 64*124
  unsigned short* hswbf  = (unsigned short*)(ws + 32768);  // 32 KB
  float*          scores = (float*)(ws + 65536);           // R*N f32 (L2-resident)
  const size_t soff = 65536 + (size_t)R * N * sizeof(float);
  int*            colIdx = (int*)(ws + soff);
  float*          lseCol = (float*)(ws + soff + (size_t)N * sizeof(float));
  float*          out    = (float*)d_out;

  k_init <<<1, 1, 0, stream>>>(sums);
  k0_hsw <<<R, 256, 0, stream>>>(hs, W, hswbf);
  k1_gemm<<<(N + 127) / 128, 256, 0, stream>>>(es, hswbf, scores, sums, N);
  k2_stats<<<1, 1, 0, stream>>>(sums, stats, N);
  k3_col <<<(N + 255) / 256, 256, 0, stream>>>(scores, mask, stats, colIdx, lseCol, N);
  k4_row <<<R, 256, 0, stream>>>(scores, mask, stats, lseRow, mMaxA, mSumA, N);
  k5_topk<<<R, 256, 0, stream>>>(scores, mask, colIdx, stats, mMaxA, mSumA,
                                 topIdx, out, N, K);
  k6_expsc<<<(R * K * R + 255) / 256, 256, 0, stream>>>(scores, stats, lseRow, lseCol,
                                                        topIdx, out + R * K, N, K);
}
